// MultiScaleEncoder_71176198029446
// MI455X (gfx1250) — matrix operand order repaired and checked by reading, not compiled, not run on hardware
//
#include <hip/hip_runtime.h>
#include <hip/hip_bf16.h>

typedef __attribute__((ext_vector_type(16))) __bf16 v16bf;
typedef __attribute__((ext_vector_type(8)))  float  v8f;

#define DEVINL __device__ __forceinline__

// ---------------------------------------------------------------------------
// FPS: one block per batch. Each of 256 threads owns N/256 = 16 consecutive
// points (blocked so argmax tie-break == lowest global index, like jnp.argmax).
// ---------------------------------------------------------------------------
__global__ void fps_kernel(const float* __restrict__ x, int* __restrict__ outidx,
                           int N, int m)
{
    const int P = 16;            // N / 256, N == 4096
    __shared__ float swd[8];
    __shared__ int   swi[8];
    __shared__ int   sfar;

    int b = blockIdx.x;
    int t = threadIdx.x;
    const float* xb = x + (size_t)b * N * 6;

    float dist[P], px[P], py[P], pz[P];
    #pragma unroll
    for (int i = 0; i < P; ++i) {
        int n = t * P + i;
        dist[i] = 1e10f;
        px[i] = xb[n * 6 + 0];
        py[i] = xb[n * 6 + 1];
        pz[i] = xb[n * 6 + 2];
    }

    int far = 0;
    if (t == 0) outidx[(size_t)b * m + 0] = 0;

    for (int it = 1; it < m; ++it) {
        float cx = xb[far * 6 + 0];
        float cy = xb[far * 6 + 1];
        float cz = xb[far * 6 + 2];
        float bd = -1.0f; int bi = 0;
        #pragma unroll
        for (int i = 0; i < P; ++i) {
            float dx = px[i] - cx, dy = py[i] - cy, dz = pz[i] - cz;
            float d = dx * dx + dy * dy + dz * dz;
            dist[i] = fminf(dist[i], d);
            if (dist[i] > bd) { bd = dist[i]; bi = t * P + i; }
        }
        #pragma unroll
        for (int off = 16; off > 0; off >>= 1) {
            float od = __shfl_down(bd, off, 32);
            int   oi = __shfl_down(bi, off, 32);
            if (od > bd || (od == bd && oi < bi)) { bd = od; bi = oi; }
        }
        if ((t & 31) == 0) { swd[t >> 5] = bd; swi[t >> 5] = bi; }
        __syncthreads();
        if (t == 0) {
            float gbd = swd[0]; int gbi = swi[0];
            #pragma unroll
            for (int w = 1; w < 8; ++w)
                if (swd[w] > gbd || (swd[w] == gbd && swi[w] < gbi)) { gbd = swd[w]; gbi = swi[w]; }
            sfar = gbi;
            outidx[(size_t)b * m + it] = gbi;
        }
        __syncthreads();
        far = sfar;
        __syncthreads();
    }
}

// ---------------------------------------------------------------------------
// Gather sampled points: xs[b,s,:] = x[b, idx[b,s], :]
// ---------------------------------------------------------------------------
__global__ void gather_xs_kernel(const float* __restrict__ x, const int* __restrict__ idx,
                                 float* __restrict__ xs, int N, int S, int B)
{
    int t = blockIdx.x * blockDim.x + threadIdx.x;
    int total = B * S * 6;
    if (t >= total) return;
    int c = t % 6;
    int s = (t / 6) % S;
    int b = t / (6 * S);
    xs[t] = x[((size_t)b * N + idx[(size_t)b * S + s]) * 6 + c];
}

// ---------------------------------------------------------------------------
// kNN on xyz: k-pass min selection with lexicographic (d, idx) ordering.
// ---------------------------------------------------------------------------
__global__ void knn_kernel(const float* __restrict__ xs, int* __restrict__ knn,
                           int S, int k, int B)
{
    int q = blockIdx.x * blockDim.x + threadIdx.x;
    if (q >= B * S) return;
    int b = q / S, s = q % S;
    const float* xb = xs + (size_t)b * S * 6;
    float qx = xb[s * 6 + 0], qy = xb[s * 6 + 1], qz = xb[s * 6 + 2];

    float lastd = -1.0f; int lasti = -1;
    for (int j = 0; j < k; ++j) {
        float bd = 3.4e38f; int bi = S;
        for (int c = 0; c < S; ++c) {
            float dx = xb[c * 6 + 0] - qx;
            float dy = xb[c * 6 + 1] - qy;
            float dz = xb[c * 6 + 2] - qz;
            float d = dx * dx + dy * dy + dz * dz;
            bool gt = (d > lastd) || (d == lastd && c > lasti);
            if (gt && (d < bd || (d == bd && c < bi))) { bd = d; bi = c; }
        }
        knn[(size_t)q * k + j] = bi;
        lastd = bd; lasti = bi;
    }
}

// ---------------------------------------------------------------------------
// Weight packing: Wp[n][Kp] (bf16, K contiguous, zero-padded to Kp) from
// row-major f32 W[K][N].  Makes every WMMA B-fragment a single aligned
// 32-byte contiguous load.
// ---------------------------------------------------------------------------
__global__ void pack_w_kernel(const float* __restrict__ W, __bf16* __restrict__ Wp,
                              int K, int N, int Kp)
{
    int k = blockIdx.x * blockDim.x + threadIdx.x;
    int n = blockIdx.y;
    if (k >= Kp) return;
    Wp[(size_t)n * Kp + k] = (k < K) ? (__bf16)W[(size_t)k * N + n] : (__bf16)0.0f;
}

// ---------------------------------------------------------------------------
// WMMA fragment layouts used here (v_wmma_f32_16x16x32_bf16, wave32; A and B share contiguous k order):
//   A 16x32 bf16 : row M = lane&15 ; element e -> K = 16*hi + e
//                  => per lane: two contiguous 8-float runs at kk+16*hi, kk+16*hi+8
//   B 32x16 bf16 : col N = lane&15 ; element e -> K = 16*hi + e
//                  => per lane: 16 contiguous K of one column (packed Wp load)
//   C/D 16x16 f32: element r -> M = r + 8*hi, N = lane&15
// ---------------------------------------------------------------------------

DEVINL void load_seg8(const float* __restrict__ ctr, const float* __restrict__ nbr,
                      int Ka, int Cin, float* seg)
{
    if (Ka < Cin) {
        const float4* p = (const float4*)(ctr + Ka);
        float4 u = p[0], v = p[1];
        seg[0] = u.x; seg[1] = u.y; seg[2] = u.z; seg[3] = u.w;
        seg[4] = v.x; seg[5] = v.y; seg[6] = v.z; seg[7] = v.w;
    } else {
        int o = Ka - Cin;
        const float4* pn = (const float4*)(nbr + o);
        const float4* pc = (const float4*)(ctr + o);
        float4 un = pn[0], vn = pn[1], uc = pc[0], vc = pc[1];
        seg[0] = un.x - uc.x; seg[1] = un.y - uc.y; seg[2] = un.z - uc.z; seg[3] = un.w - uc.w;
        seg[4] = vn.x - vc.x; seg[5] = vn.y - vc.y; seg[6] = vn.z - vc.z; seg[7] = vn.w - vc.w;
    }
}

// Edge conv: one wave per (b, s, 64-wide output strip) = 4 WMMA tiles.
// out[b,s,n] = max_j relu( [ctr, nbr_j - ctr] @ W + bias )[n]
__global__ void edgeconv_wmma(const float* __restrict__ fin, const int* __restrict__ knn,
                              const __bf16* __restrict__ Wp, const float* __restrict__ bias,
                              float* __restrict__ fout,
                              int S, int Cin, int k, int Cout, int Kp,
                              int ntgShift, int sShift)
{
    int bid = blockIdx.x;
    int tg = bid & ((1 << ntgShift) - 1);
    int s  = (bid >> ntgShift) & ((1 << sShift) - 1);
    int b  =  bid >> (ntgShift + sShift);
    int n0 = tg << 6;
    int lane = threadIdx.x;
    int l  = lane & 15;
    int hi = lane >> 4;
    int K2 = 2 * Cin;

    const float* ctr = fin + ((size_t)b * S + s) * Cin;
    const int*   nb  = knn + ((size_t)b * S + s) * k;

    const __bf16* w0 = Wp + (size_t)(n0 +  0 + l) * Kp + 16 * hi;
    const __bf16* w1 = Wp + (size_t)(n0 + 16 + l) * Kp + 16 * hi;
    const __bf16* w2 = Wp + (size_t)(n0 + 32 + l) * Kp + 16 * hi;
    const __bf16* w3 = Wp + (size_t)(n0 + 48 + l) * Kp + 16 * hi;

    float bv0 = bias[n0 +  0 + l];
    float bv1 = bias[n0 + 16 + l];
    float bv2 = bias[n0 + 32 + l];
    float bv3 = bias[n0 + 48 + l];

    float rm0 = -3.0e38f, rm1 = -3.0e38f, rm2 = -3.0e38f, rm3 = -3.0e38f;

    for (int j0 = 0; j0 < k; j0 += 16) {
        v8f a0 = {}, a1 = {}, a2 = {}, a3 = {};
        int j = nb[j0 + l];                      // neighbor of row (lane&15)
        const float* nbr = fin + ((size_t)b * S + j) * Cin;
        __builtin_prefetch(nbr, 0, 3);

        for (int kk = 0; kk < K2; kk += 32) {
            // prefetch next chunk of the weight stream while WMMAs run
            __builtin_prefetch(w0 + kk + 32, 0, 3);
            __builtin_prefetch(w1 + kk + 32, 0, 3);
            __builtin_prefetch(w2 + kk + 32, 0, 3);
            __builtin_prefetch(w3 + kk + 32, 0, 3);

            v16bf av;
            if (Cin >= 16) {
                float seg[16];
                load_seg8(ctr, nbr, kk + 16 * hi,     Cin, seg);
                load_seg8(ctr, nbr, kk + 16 * hi + 8, Cin, seg + 8);
                #pragma unroll
                for (int e = 0; e < 16; ++e) av[e] = (__bf16)seg[e];
            } else {
                #pragma unroll
                for (int e = 0; e < 16; ++e) {
                    int Ka = kk + 16 * hi + e;
                    float va = 0.0f;
                    if (Ka < K2)
                        va = (Ka < Cin) ? ctr[Ka] : (nbr[Ka - Cin] - ctr[Ka - Cin]);
                    av[e] = (__bf16)va;
                }
            }
            v16bf b0 = *(const v16bf*)(w0 + kk);
            v16bf b1 = *(const v16bf*)(w1 + kk);
            v16bf b2 = *(const v16bf*)(w2 + kk);
            v16bf b3 = *(const v16bf*)(w3 + kk);
            a0 = __builtin_amdgcn_wmma_f32_16x16x32_bf16(false, av, false, b0, (short)0, a0, false, false);
            a1 = __builtin_amdgcn_wmma_f32_16x16x32_bf16(false, av, false, b1, (short)0, a1, false, false);
            a2 = __builtin_amdgcn_wmma_f32_16x16x32_bf16(false, av, false, b2, (short)0, a2, false, false);
            a3 = __builtin_amdgcn_wmma_f32_16x16x32_bf16(false, av, false, b3, (short)0, a3, false, false);
        }
        // bias + relu + max over 16 rows (neighbors) of this chunk, per tile
        float p0 = -3.0e38f, p1 = -3.0e38f, p2 = -3.0e38f, p3 = -3.0e38f;
        #pragma unroll
        for (int r = 0; r < 8; ++r) {
            p0 = fmaxf(p0, fmaxf(a0[r] + bv0, 0.0f));
            p1 = fmaxf(p1, fmaxf(a1[r] + bv1, 0.0f));
            p2 = fmaxf(p2, fmaxf(a2[r] + bv2, 0.0f));
            p3 = fmaxf(p3, fmaxf(a3[r] + bv3, 0.0f));
        }
        p0 = fmaxf(p0, __shfl_xor(p0, 16, 32));
        p1 = fmaxf(p1, __shfl_xor(p1, 16, 32));
        p2 = fmaxf(p2, __shfl_xor(p2, 16, 32));
        p3 = fmaxf(p3, __shfl_xor(p3, 16, 32));
        rm0 = fmaxf(rm0, p0); rm1 = fmaxf(rm1, p1);
        rm2 = fmaxf(rm2, p2); rm3 = fmaxf(rm3, p3);
    }
    if (hi == 0) {
        float* orow = fout + ((size_t)b * S + s) * Cout + n0;
        orow[ 0 + l] = rm0;
        orow[16 + l] = rm1;
        orow[32 + l] = rm2;
        orow[48 + l] = rm3;
    }
}

// Plain GEMM + bias: out[M,N] = A[M,K] @ W[K,N] + bias.
// One wave per 16x64 strip (4 WMMA tiles). Requires K % 32 == 0.
__global__ void gemm_wmma(const float* __restrict__ A, const __bf16* __restrict__ Wp,
                          const float* __restrict__ bias, float* __restrict__ out,
                          int K, int N, int ntgShift)
{
    int bid = blockIdx.x;
    int tg = bid & ((1 << ntgShift) - 1);
    int m0 = (bid >> ntgShift) << 4;
    int n0 = tg << 6;
    int lane = threadIdx.x;
    int l  = lane & 15;
    int hi = lane >> 4;

    const float* arow = A + (size_t)(m0 + l) * K;
    const __bf16* w0 = Wp + (size_t)(n0 +  0 + l) * K + 16 * hi;
    const __bf16* w1 = Wp + (size_t)(n0 + 16 + l) * K + 16 * hi;
    const __bf16* w2 = Wp + (size_t)(n0 + 32 + l) * K + 16 * hi;
    const __bf16* w3 = Wp + (size_t)(n0 + 48 + l) * K + 16 * hi;

    v8f a0 = {}, a1 = {}, a2 = {}, a3 = {};

    #pragma unroll 2
    for (int kk = 0; kk < K; kk += 32) {
        __builtin_prefetch(arow + kk + 64, 0, 3);
        __builtin_prefetch(w0 + kk + 32, 0, 3);
        __builtin_prefetch(w1 + kk + 32, 0, 3);
        __builtin_prefetch(w2 + kk + 32, 0, 3);
        __builtin_prefetch(w3 + kk + 32, 0, 3);

        const float4* p0 = (const float4*)(arow + kk + 16 * hi);
        const float4* p1 = (const float4*)(arow + kk + 16 * hi + 8);
        float4 u0 = p0[0], u1 = p0[1], u2 = p1[0], u3 = p1[1];
        v16bf av;
        av[ 0] = (__bf16)u0.x; av[ 1] = (__bf16)u0.y; av[ 2] = (__bf16)u0.z; av[ 3] = (__bf16)u0.w;
        av[ 4] = (__bf16)u1.x; av[ 5] = (__bf16)u1.y; av[ 6] = (__bf16)u1.z; av[ 7] = (__bf16)u1.w;
        av[ 8] = (__bf16)u2.x; av[ 9] = (__bf16)u2.y; av[10] = (__bf16)u2.z; av[11] = (__bf16)u2.w;
        av[12] = (__bf16)u3.x; av[13] = (__bf16)u3.y; av[14] = (__bf16)u3.z; av[15] = (__bf16)u3.w;

        v16bf b0 = *(const v16bf*)(w0 + kk);
        v16bf b1 = *(const v16bf*)(w1 + kk);
        v16bf b2 = *(const v16bf*)(w2 + kk);
        v16bf b3 = *(const v16bf*)(w3 + kk);
        a0 = __builtin_amdgcn_wmma_f32_16x16x32_bf16(false, av, false, b0, (short)0, a0, false, false);
        a1 = __builtin_amdgcn_wmma_f32_16x16x32_bf16(false, av, false, b1, (short)0, a1, false, false);
        a2 = __builtin_amdgcn_wmma_f32_16x16x32_bf16(false, av, false, b2, (short)0, a2, false, false);
        a3 = __builtin_amdgcn_wmma_f32_16x16x32_bf16(false, av, false, b3, (short)0, a3, false, false);
    }
    float bv0 = bias[n0 +  0 + l];
    float bv1 = bias[n0 + 16 + l];
    float bv2 = bias[n0 + 32 + l];
    float bv3 = bias[n0 + 48 + l];
    #pragma unroll
    for (int r = 0; r < 8; ++r) {
        float* orow = out + (size_t)(m0 + r + 8 * hi) * N + n0;
        orow[ 0 + l] = a0[r] + bv0;
        orow[16 + l] = a1[r] + bv1;
        orow[32 + l] = a2[r] + bv2;
        orow[48 + l] = a3[r] + bv3;
    }
}

// ---------------------------------------------------------------------------
// 3-NN inverse-distance upsample.
// ---------------------------------------------------------------------------
__global__ void up3_kernel(const float* __restrict__ x, const float* __restrict__ xs,
                           float* __restrict__ w3, int* __restrict__ i3,
                           int N, int S, int B)
{
    int q = blockIdx.x * blockDim.x + threadIdx.x;
    if (q >= B * N) return;
    int b = q / N, n = q % N;
    const float* xp = x + ((size_t)b * N + n) * 6;
    const float* xb = xs + (size_t)b * S * 6;
    float qx = xp[0], qy = xp[1], qz = xp[2];

    float d0 = 3.4e38f, d1 = 3.4e38f, d2 = 3.4e38f;
    int   i0 = 0,       i1 = 0,       i2 = 0;
    for (int c = 0; c < S; ++c) {
        float dx = xb[c * 6 + 0] - qx;
        float dy = xb[c * 6 + 1] - qy;
        float dz = xb[c * 6 + 2] - qz;
        float d = sqrtf(fmaxf(dx * dx + dy * dy + dz * dz, 0.0f));
        if (d < d0)      { d2 = d1; i2 = i1; d1 = d0; i1 = i0; d0 = d; i0 = c; }
        else if (d < d1) { d2 = d1; i2 = i1; d1 = d;  i1 = c; }
        else if (d < d2) { d2 = d;  i2 = c; }
    }
    float w0 = 1.0f / (d0 + 1e-8f);
    float w1 = 1.0f / (d1 + 1e-8f);
    float w2 = 1.0f / (d2 + 1e-8f);
    float inv = 1.0f / (w0 + w1 + w2);
    w3[(size_t)q * 3 + 0] = w0 * inv;
    w3[(size_t)q * 3 + 1] = w1 * inv;
    w3[(size_t)q * 3 + 2] = w2 * inv;
    i3[(size_t)q * 3 + 0] = i0;
    i3[(size_t)q * 3 + 1] = i1;
    i3[(size_t)q * 3 + 2] = i2;
}

__global__ void up_apply_kernel(const float* __restrict__ fs, const float* __restrict__ w3,
                                const int* __restrict__ i3, float* __restrict__ fused,
                                int N, int S, int C, int coff, int CF, int B)
{
    int t = blockIdx.x * blockDim.x + threadIdx.x;
    int total = B * N * C;
    if (t >= total) return;
    int c = t % C;
    int n = (t / C) % N;
    int b = t / (C * N);
    int q = b * N + n;
    const float* fb = fs + (size_t)b * S * C;
    float v = w3[(size_t)q * 3 + 0] * fb[(size_t)i3[(size_t)q * 3 + 0] * C + c]
            + w3[(size_t)q * 3 + 1] * fb[(size_t)i3[(size_t)q * 3 + 1] * C + c]
            + w3[(size_t)q * 3 + 2] * fb[(size_t)i3[(size_t)q * 3 + 2] * C + c];
    fused[((size_t)b * N + n) * CF + coff + c] = v;
}

// ---------------------------------------------------------------------------
// GroupNorm(32) + ReLU in-place; optional fused sinusoidal positional encoding.
// ---------------------------------------------------------------------------
__global__ void gn_relu_pe_kernel(float* __restrict__ h, const float* __restrict__ gw,
                                  const float* __restrict__ gb, int rows, int C,
                                  int addpe, int N)
{
    const int G = 32;
    const int CG = 16;   // C / G, C == 512
    int t = blockIdx.x * blockDim.x + threadIdx.x;
    if (t >= rows * G) return;
    int grp = t % G;
    int row = t / G;
    float* p = h + (size_t)row * C + grp * CG;

    float v[CG];
    float mu = 0.0f;
    #pragma unroll
    for (int i = 0; i < CG; ++i) { v[i] = p[i]; mu += v[i]; }
    mu *= (1.0f / CG);
    float var = 0.0f;
    #pragma unroll
    for (int i = 0; i < CG; ++i) { float d = v[i] - mu; var += d * d; }
    var *= (1.0f / CG);
    float rs = rsqrtf(var + 1e-5f);

    int n = row % N;
    #pragma unroll
    for (int i = 0; i < CG; ++i) {
        int c = grp * CG + i;
        float y = (v[i] - mu) * rs * gw[c] + gb[c];
        y = fmaxf(y, 0.0f);
        if (addpe) {
            int ii = c >> 1;
            float ang = (float)n * expf(-(float)(2 * ii) * 0.017988946f);  // ln(1e4)/512
            y += (c & 1) ? cosf(ang) : sinf(ang);
        }
        p[i] = y;
    }
}

// ---------------------------------------------------------------------------
// Host-side drivers
// ---------------------------------------------------------------------------
static inline int ilog2(int v) { return 31 - __builtin_clz(v); }

static void pack_weights(const float* W, __bf16* Wp, int K, int N, int Kp,
                         hipStream_t stream)
{
    dim3 g((Kp + 255) / 256, N);
    pack_w_kernel<<<g, 256, 0, stream>>>(W, Wp, K, N, Kp);
}

static void run_encoder(const float* x, int B, int N, int S, int k,
                        int nlayers, const float* const* Ws, const float* const* bs,
                        const int* Cs,
                        const float* Wo, const float* bo, int Cout,
                        float* xs, int* fpsidx, int* knnb,
                        float* fA, float* fB, float* fC, __bf16* Wpack,
                        float* w3, int* i3, float* fused, int coff, int CF,
                        hipStream_t stream)
{
    fps_kernel<<<B, 256, 0, stream>>>(x, fpsidx, N, S);

    int tot = B * S * 6;
    gather_xs_kernel<<<(tot + 255) / 256, 256, 0, stream>>>(x, fpsidx, xs, N, S, B);

    knn_kernel<<<(B * S + 127) / 128, 128, 0, stream>>>(xs, knnb, S, k, B);

    const float* fin = xs;
    int Cin = 6;
    float* bufs[2] = { fA, fB };
    int sShift = ilog2(S);
    for (int L = 0; L < nlayers; ++L) {
        int Co = Cs[L];
        int K2 = 2 * Cin;
        int Kp = (K2 + 31) & ~31;
        pack_weights(Ws[L], Wpack, K2, Co, Kp, stream);
        float* fo = bufs[L & 1];
        int ntgShift = ilog2(Co >> 6);
        int blocks = B * S * (Co >> 6);
        edgeconv_wmma<<<blocks, 32, 0, stream>>>(fin, knnb, Wpack, bs[L], fo,
                                                 S, Cin, k, Co, Kp, ntgShift, sShift);
        fin = fo;
        Cin = Co;
    }

    // fout = fin @ Wo + bo   ([B*S, Cin] x [Cin, Cout]); Cin % 32 == 0 here
    pack_weights(Wo, Wpack, Cin, Cout, Cin, stream);
    gemm_wmma<<<(B * S / 16) * (Cout >> 6), 32, 0, stream>>>(
        fin, Wpack, bo, fC, Cin, Cout, ilog2(Cout >> 6));

    up3_kernel<<<(B * N + 127) / 128, 128, 0, stream>>>(x, xs, w3, i3, N, S, B);

    int tot2 = B * N * Cout;
    up_apply_kernel<<<(tot2 + 255) / 256, 256, 0, stream>>>(
        fC, w3, i3, fused, N, S, Cout, coff, CF, B);
}

extern "C" void kernel_launch(void* const* d_in, const int* in_sizes, int n_in,
                              void* d_out, int out_size, void* d_ws, size_t ws_size,
                              hipStream_t stream)
{
    (void)in_sizes; (void)n_in; (void)out_size; (void)ws_size;

    const int B = 2, N = 4096, CF = 1280, C = 512;

    const float* x = (const float*)d_in[0];

    // ---- workspace layout ----
    float* fused  = (float*)d_ws;                          // B*N*CF     = 10,485,760 f
    float* h1     = fused + (size_t)B * N * CF;            // B*N*512    =  4,194,304 f
    float* fA     = h1    + (size_t)B * N * C;             // B*2048*512 =  2,097,152 f
    float* fB     = fA    + (size_t)B * 2048 * 512;
    float* fC     = fB    + (size_t)B * 2048 * 512;
    float* xs     = fC    + (size_t)B * 2048 * 512;        // B*2048*6
    float* w3     = xs    + (size_t)B * 2048 * 6;          // B*N*3
    int*   i3     = (int*)(w3 + (size_t)B * N * 3);        // B*N*3
    int*   fpsidx = i3    + (size_t)B * N * 3;             // B*2048
    int*   knnb   = fpsidx + (size_t)B * 2048;             // B*2048*64
    __bf16* Wpack = (__bf16*)(knnb + (size_t)B * 2048 * 64); // up to 1280*512 bf16

    // ---- encoder 1: S=512, k=16, 6->128->256, Wo 256->256, coff 0 ----
    {
        const float* Ws[2] = { (const float*)d_in[1], (const float*)d_in[3] };
        const float* bs[2] = { (const float*)d_in[2], (const float*)d_in[4] };
        int Cs[2] = { 128, 256 };
        run_encoder(x, B, N, 512, 16, 2, Ws, bs, Cs,
                    (const float*)d_in[5], (const float*)d_in[6], 256,
                    xs, fpsidx, knnb, fA, fB, fC, Wpack, w3, i3, fused, 0, CF, stream);
    }
    // ---- encoder 2: S=1024, k=32, 6->256->512, Wo 512->512, coff 256 ----
    {
        const float* Ws[2] = { (const float*)d_in[7], (const float*)d_in[9] };
        const float* bs[2] = { (const float*)d_in[8], (const float*)d_in[10] };
        int Cs[2] = { 256, 512 };
        run_encoder(x, B, N, 1024, 32, 2, Ws, bs, Cs,
                    (const float*)d_in[11], (const float*)d_in[12], 512,
                    xs, fpsidx, knnb, fA, fB, fC, Wpack, w3, i3, fused, 256, CF, stream);
    }
    // ---- encoder 3: S=2048, k=64, 6->128->256->512, Wo 512->512, coff 768 ----
    {
        const float* Ws[3] = { (const float*)d_in[13], (const float*)d_in[15], (const float*)d_in[17] };
        const float* bs[3] = { (const float*)d_in[14], (const float*)d_in[16], (const float*)d_in[18] };
        int Cs[3] = { 128, 256, 512 };
        run_encoder(x, B, N, 2048, 64, 3, Ws, bs, Cs,
                    (const float*)d_in[19], (const float*)d_in[20], 512,
                    xs, fpsidx, knnb, fA, fB, fC, Wpack, w3, i3, fused, 768, CF, stream);
    }

    // ---- fusion ----
    const float* fW1 = (const float*)d_in[21];
    const float* fb1 = (const float*)d_in[22];
    const float* g1w = (const float*)d_in[23];
    const float* g1b = (const float*)d_in[24];
    const float* fW2 = (const float*)d_in[25];
    const float* fb2 = (const float*)d_in[26];
    const float* g2w = (const float*)d_in[27];
    const float* g2b = (const float*)d_in[28];

    int Mrows = B * N;  // 8192
    float* out = (float*)d_out;

    pack_weights(fW1, Wpack, CF, C, CF, stream);
    gemm_wmma<<<(Mrows / 16) * (C >> 6), 32, 0, stream>>>(
        fused, Wpack, fb1, h1, CF, C, ilog2(C >> 6));
    gn_relu_pe_kernel<<<(Mrows * 32 + 255) / 256, 256, 0, stream>>>(
        h1, g1w, g1b, Mrows, C, 0, N);

    pack_weights(fW2, Wpack, C, C, C, stream);
    gemm_wmma<<<(Mrows / 16) * (C >> 6), 32, 0, stream>>>(
        h1, Wpack, fb2, out, C, C, ilog2(C >> 6));
    gn_relu_pe_kernel<<<(Mrows * 32 + 255) / 256, 256, 0, stream>>>(
        out, g2w, g2b, Mrows, C, 1, N);
}
